// SMEFTNet_23338852286866
// MI455X (gfx1250) — matrix operand order repaired and checked by reading, not compiled, not run on hardware
//
#include <hip/hip_runtime.h>
#include <hip/hip_bf16.h>

// SMEFTNet on gfx1250 (MI455X, wave32).
// Pipeline: prep -> pair(L0) -> aggGEMM(L0,WMMA) -> finalize(L0)
//        -> preGEMM(L1,WMMA) -> pair(L1) -> aggGEMM(L1,WMMA) -> finalize(L1) -> readout
// All GEMMs use v_wmma_f32_16x16x32_f16 (f16 operands, f32 accumulate).

typedef __attribute__((ext_vector_type(16))) _Float16 v16h;
typedef __attribute__((ext_vector_type(8)))  float    v8f;

#define B_   16
#define N_   128
#define H_   64
#define NEG_ 0.01f
#define DRN2_ 0.16f           // 0.4^2
#define PI2_ 6.283185307179586f

__device__ __forceinline__ float leaky1(float z){ return z >= 0.f ? z : NEG_*z; }

// ---- WMMA fragment loaders (contiguous k order, same for A and B) ------------------------
// A (16x32 f16): lanes 0-15 -> M=lane, K = 0..15; lanes 16-31 -> M=lane-16, K = 16..31
__device__ __forceinline__ v16h load_a_f32(const float* __restrict__ A, int ldA, int k0, int lane) {
  int m = lane & 15, kh = lane >> 4;
  const float* rp = A + (size_t)m * ldA + k0 + kh * 16;
  v16h a;
#pragma unroll
  for (int c = 0; c < 16; ++c) {
    a[c]     = (_Float16)rp[c];
  }
  return a;
}

// B (32x16 f16, row-major k x n in memory): lanes 0-15 -> N=lane, K=0..15; lanes 16-31 -> K=16..31
__device__ __forceinline__ v16h load_b_h16(const _Float16* __restrict__ Bm, int ldB, int k0, int n0, int lane) {
  int n = n0 + (lane & 15), kg = lane >> 4;
  const _Float16* bp = Bm + (size_t)(k0 + kg * 16) * ldB + n;
  v16h b;
#pragma unroll
  for (int c = 0; c < 16; ++c) b[c] = bp[(size_t)c * ldB];
  return b;
}

// B from f32 with column guard (for the 64x65 projection weights)
__device__ __forceinline__ v16h load_b_f32_guard(const float* __restrict__ Bm, int ldB, int ncols,
                                                 int k0, int n0, int lane) {
  int n = n0 + (lane & 15), kg = lane >> 4;
  v16h b;
#pragma unroll
  for (int c = 0; c < 16; ++c)
    b[c] = (n < ncols) ? (_Float16)Bm[(size_t)(k0 + kg * 16 + c) * ldB + n] : (_Float16)0.f;
  return b;
}

// ---- Kernels -----------------------------------------------------------------

// Fold layer-0 weights into per-node pre-activations, derive layer-1 folded f16 weights.
__global__ void prep_kernel(const float* __restrict__ angles, const float* __restrict__ W00,
                            const float* __restrict__ W10,
                            float* __restrict__ preA, float* __restrict__ preB,
                            _Float16* __restrict__ WA1h, _Float16* __restrict__ WB1h) {
  int idx = blockIdx.x * blockDim.x + threadIdx.x;
  if (idx < B_ * N_ * H_) {
    int bn = idx >> 6, h = idx & 63;
    float ax = angles[bn * 2], ay = angles[bn * 2 + 1];
    float r = sqrtf(ax * ax + ay * ay);                 // layer-0 feature f = r (Df=1)
    preA[idx] = r * (W00[0 * H_ + h] - W00[2 * H_ + h]);  // fi@(Wa - Wc)
    preB[idx] = r * (W00[1 * H_ + h] + W00[2 * H_ + h]);  // fj@(Wb + Wc)
  }
  if (idx < H_ * H_) {
    int k = idx >> 6, h = idx & 63;
    WA1h[idx] = (_Float16)(W10[k * H_ + h]            - W10[(2 * H_ + k) * H_ + h]);
    WB1h[idx] = (_Float16)(W10[(H_ + k) * H_ + h]     + W10[(2 * H_ + k) * H_ + h]);
  }
}

// One wave32 per node i: fused adjacency + cos/sin + leaky MLP-hidden + weighted aggregation over j.
// Lane l handles hidden channels {l, l+32}. 4 waves (4 i's) per block, one batch per 32 blocks.
__global__ void pair_kernel(const float* __restrict__ ptc, const float* __restrict__ a,
                            const float* __restrict__ preA, const float* __restrict__ preB,
                            const float* __restrict__ wc, const float* __restrict__ ws,
                            const float* __restrict__ bias,
                            float* __restrict__ aggh, float* __restrict__ s_out) {
  __shared__ float sax[N_], say[N_], sr[N_], spt[N_];
  int b = blockIdx.x >> 5;
  int t = threadIdx.x;
  {
    float ax = a[(b * N_ + t) * 2], ay = a[(b * N_ + t) * 2 + 1];
    sax[t] = ax; say[t] = ay;
    sr[t]  = sqrtf(ax * ax + ay * ay);
    spt[t] = ptc[b * N_ + t];
  }
  __syncthreads();
  int wid = t >> 5, lane = t & 31;
  int i  = ((blockIdx.x & 31) << 2) + wid;
  int bi = b * N_ + i;
  float axi = sax[i], ayi = say[i], ri = sr[i];
  float p0 = preA[(size_t)bi * H_ + lane], p1 = preA[(size_t)bi * H_ + lane + 32];
  float wc0 = wc[lane], wc1 = wc[lane + 32];
  float ws0 = ws[lane], ws1 = ws[lane + 32];
  float bb0 = bias[lane], bb1 = bias[lane + 32];
  float acc0 = 0.f, acc1 = 0.f, wsum = 0.f;
  for (int j = 0; j < N_; ++j) {
    float dx = axi - sax[j], dy = ayi - say[j];
    float d2 = dx * dx + dy * dy;
    if (d2 <= DRN2_) {                       // uniform across the wave (same i,j)
      float inv = 1.f / (ri * sr[j]);        // norm = sqrt(r2_i * r2_j)
      float cs = (axi * sax[j] + ayi * say[j]) * inv;
      float sn = (ayi * sax[j] - axi * say[j]) * inv;
      const float* pjr = preB + (size_t)(b * N_ + j) * H_;
      float pj = spt[j];                     // weight = adj * ptc_j
      float z0 = p0 + pjr[lane]      + cs * wc0 + sn * ws0 + bb0;
      float z1 = p1 + pjr[lane + 32] + cs * wc1 + sn * ws1 + bb1;
      acc0 += pj * leaky1(z0);
      acc1 += pj * leaky1(z1);
      wsum += pj;
    }
  }
  float invd = 1.f / wsum;                   // denom = sum_j adj*ptc_j
  aggh[(size_t)bi * H_ + lane]      = acc0 * invd;
  aggh[(size_t)bi * H_ + lane + 32] = acc1 * invd;
  if (lane == 0) s_out[bi] = wsum * invd;    // s = sum_j wj
}

// preA/preB = F(2048x64, f32 src with stride ldF) @ WA1h/WB1h (64x64 f16). One wave per 16x16 tile.
__global__ void pre_gemm_kernel(const float* __restrict__ F, int ldF,
                                const _Float16* __restrict__ WA, const _Float16* __restrict__ WB,
                                float* __restrict__ preA, float* __restrict__ preB) {
  int w = blockIdx.x * (blockDim.x >> 5) + (threadIdx.x >> 5);
  int lane = threadIdx.x & 31;
  int wt  = w >> 9;                 // 0 -> WA/preA, 1 -> WB/preB
  int rem = w & 511;
  int tn  = rem >> 7;               // 0..3
  int tm  = rem & 127;              // 0..127
  const _Float16* Bm = wt ? WB : WA;
  float* Out = wt ? preB : preA;
  v8f c = {};
#pragma unroll
  for (int ks = 0; ks < 2; ++ks) {
    v16h af = load_a_f32(F + (size_t)(tm * 16) * ldF, ldF, ks * 32, lane);
    v16h bf = load_b_h16(Bm, H_, ks * 32, tn * 16, lane);
    c = __builtin_amdgcn_wmma_f32_16x16x32_f16(false, af, false, bf, (short)0, c, false, false);
  }
  int n = tn * 16 + (lane & 15);
  int mh = (lane >> 4) * 8;
#pragma unroll
  for (int r2 = 0; r2 < 8; ++r2)
    Out[(size_t)(tm * 16 + mh + r2) * H_ + n] = c[r2];
}

// aggm(2048x65) = aggh(2048x64) @ W2(64x65) + s*b2. One wave per 16x16 tile, 5 col-tiles (guarded).
__global__ void agg_gemm_kernel(const float* __restrict__ aggh, const float* __restrict__ W2,
                                const float* __restrict__ b2, const float* __restrict__ s,
                                float* __restrict__ aggm) {
  int w = blockIdx.x * (blockDim.x >> 5) + (threadIdx.x >> 5);
  int lane = threadIdx.x & 31;
  int tn = w / 128;                 // 0..4
  int tm = w % 128;
  v8f c = {};
#pragma unroll
  for (int ks = 0; ks < 2; ++ks) {
    v16h af = load_a_f32(aggh + (size_t)(tm * 16) * H_, H_, ks * 32, lane);
    v16h bf = load_b_f32_guard(W2, H_ + 1, H_ + 1, ks * 32, tn * 16, lane);
    c = __builtin_amdgcn_wmma_f32_16x16x32_f16(false, af, false, bf, (short)0, c, false, false);
  }
  int n = tn * 16 + (lane & 15);
  if (n < H_ + 1) {
    float bn = b2[n];
    int mh = (lane >> 4) * 8;
#pragma unroll
    for (int r2 = 0; r2 < 8; ++r2) {
      int m = tm * 16 + mh + r2;
      aggm[(size_t)m * (H_ + 1) + n] = c[r2] + s[m] * bn;
    }
  }
}

// Per-node epilogue: agg_pt = s*ptc, rotate agg_a = s*a by 2*pi*gamma (gamma = aggm[:,64]).
__global__ void finalize_kernel(const float* __restrict__ s, const float* __restrict__ ptc_prev,
                                const float* __restrict__ a_prev, const float* __restrict__ aggm,
                                float* __restrict__ ptc_next, float* __restrict__ a_next) {
  int bn = blockIdx.x * blockDim.x + threadIdx.x;
  if (bn >= B_ * N_) return;
  float sv = s[bn];
  float ang = PI2_ * aggm[(size_t)bn * (H_ + 1) + H_];
  float co = cosf(ang), si = sinf(ang);
  float gax = sv * a_prev[bn * 2], gay = sv * a_prev[bn * 2 + 1];
  a_next[bn * 2]     = co * gax - si * gay;
  a_next[bn * 2 + 1] = si * gax + co * gay;
  ptc_next[bn] = sv * ptc_prev[bn];
}

// Global pooling + 64->32->32->1 MLP + sigmoid. One block of 64 threads per batch.
__global__ void readout_kernel(const float* __restrict__ ptc2, const float* __restrict__ aggm,
                               const float* __restrict__ a2,
                               const float* __restrict__ Wr0, const float* __restrict__ br0,
                               const float* __restrict__ Wr1, const float* __restrict__ br1,
                               const float* __restrict__ Wr2, const float* __restrict__ br2,
                               float* __restrict__ out) {
  __shared__ float red[64];
  __shared__ float gsh[66];
  __shared__ float h1[32];
  __shared__ float h2[32];
  int t = threadIdx.x, b = blockIdx.x;
  red[t] = ptc2[b * N_ + t] + ptc2[b * N_ + t + 64];
  __syncthreads();
  for (int o = 32; o > 0; o >>= 1) { if (t < o) red[t] += red[t + o]; __syncthreads(); }
  float inv_sum = 1.f / red[0];

  float acc = 0.f;
  for (int n = 0; n < N_; ++n)
    acc += ptc2[b * N_ + n] * aggm[(size_t)(b * N_ + n) * (H_ + 1) + t];
  gsh[t] = acc * inv_sum;
  if (t < 2) {
    float aa = 0.f;
    for (int n = 0; n < N_; ++n)
      aa += ptc2[b * N_ + n] * a2[(size_t)(b * N_ + n) * 2 + t];
    gsh[64 + t] = aa * inv_sum;
  }
  __syncthreads();
  if (t < 32) {
    float z = br0[t];
    for (int h = 0; h < H_; ++h) z += gsh[h] * Wr0[h * 32 + t];
    h1[t] = leaky1(z);
  }
  __syncthreads();
  if (t < 32) {
    float z = br1[t];
    for (int k = 0; k < 32; ++k) z += h1[k] * Wr1[k * 32 + t];
    h2[t] = leaky1(z);
  }
  __syncthreads();
  if (t == 0) {
    float z = br2[0];
    for (int k = 0; k < 32; ++k) z += h2[k] * Wr2[k];
    out[b * 3 + 0] = 1.f / (1.f + expf(-z));
    out[b * 3 + 1] = gsh[64];
    out[b * 3 + 2] = gsh[65];
  }
}

// ---- Host launcher -----------------------------------------------------------
extern "C" void kernel_launch(void* const* d_in, const int* in_sizes, int n_in,
                              void* d_out, int out_size, void* d_ws, size_t ws_size,
                              hipStream_t stream) {
  const float* pt     = (const float*)d_in[0];
  const float* angles = (const float*)d_in[1];
  const float* W0_0   = (const float*)d_in[2];
  const float* b0_0   = (const float*)d_in[3];
  const float* W0_1   = (const float*)d_in[4];
  const float* b0_1   = (const float*)d_in[5];
  const float* W1_0   = (const float*)d_in[6];
  const float* b1_0   = (const float*)d_in[7];
  const float* W1_1   = (const float*)d_in[8];
  const float* b1_1   = (const float*)d_in[9];
  const float* Wr0    = (const float*)d_in[10];
  const float* br0    = (const float*)d_in[11];
  const float* Wr1    = (const float*)d_in[12];
  const float* br1    = (const float*)d_in[13];
  const float* Wr2    = (const float*)d_in[14];
  const float* br2    = (const float*)d_in[15];
  float* out = (float*)d_out;

  // Workspace layout (bytes), ~2.1 MB total, buffers reused across layers.
  char* ws = (char*)d_ws;
  float*    preA  = (float*)   (ws + 0);          // 2048*64*4 = 524288
  float*    preB  = (float*)   (ws + 524288);     // 524288
  float*    aggh  = (float*)   (ws + 1048576);    // 524288
  float*    aggm  = (float*)   (ws + 1572864);    // 2048*65*4 = 532480
  float*    sbuf  = (float*)   (ws + 2105344);    // 8192
  float*    ptc1  = (float*)   (ws + 2113536);    // 8192
  float*    a1    = (float*)   (ws + 2121728);    // 16384
  float*    ptc2  = (float*)   (ws + 2138112);    // 8192
  float*    a2    = (float*)   (ws + 2146304);    // 16384
  _Float16* WA1h  = (_Float16*)(ws + 2162688);    // 8192
  _Float16* WB1h  = (_Float16*)(ws + 2170880);    // 8192

  // Layer 0 trig weights/bias live directly in W0_0 rows 3,4; layer 1 in W1_0 rows 192,193.
  const float* wc0 = W0_0 + 3 * H_;
  const float* ws0 = W0_0 + 4 * H_;
  const float* wc1 = W1_0 + 192 * H_;
  const float* ws1 = W1_0 + 193 * H_;

  prep_kernel<<<512, 256, 0, stream>>>(angles, W0_0, W1_0, preA, preB, WA1h, WB1h);

  // ---- Layer 0
  pair_kernel<<<512, 128, 0, stream>>>(pt, angles, preA, preB, wc0, ws0, b0_0, aggh, sbuf);
  agg_gemm_kernel<<<160, 128, 0, stream>>>(aggh, W0_1, b0_1, sbuf, aggm);
  finalize_kernel<<<8, 256, 0, stream>>>(sbuf, pt, angles, aggm, ptc1, a1);

  // ---- Layer 1 (f1 = aggm cols 0..63, row stride 65)
  pre_gemm_kernel<<<256, 128, 0, stream>>>(aggm, H_ + 1, WA1h, WB1h, preA, preB);
  pair_kernel<<<512, 128, 0, stream>>>(ptc1, a1, preA, preB, wc1, ws1, b1_0, aggh, sbuf);
  agg_gemm_kernel<<<160, 128, 0, stream>>>(aggh, W1_1, b1_1, sbuf, aggm);
  finalize_kernel<<<8, 256, 0, stream>>>(sbuf, ptc1, a1, aggm, ptc2, a2);

  // ---- Readout
  readout_kernel<<<16, 64, 0, stream>>>(ptc2, aggm, a2, Wr0, br0, Wr1, br1, Wr2, br2, out);
}
